// CycleMLP1w1a_78091095375879
// MI455X (gfx1250) — 8-bit matrix operand order repaired and checked by reading, compiled, not run on hardware
//
#include <hip/hip_runtime.h>
#include <hip/hip_bf16.h>
#include <stdint.h>

typedef int v8i __attribute__((ext_vector_type(8)));
typedef int v4i __attribute__((ext_vector_type(4)));

#define IN_CH 384
#define HW 32
#define NB 64
#define KTILES 6
#define MTILES 24
#define NTILES 2

__global__ void quant_weight_kernel(const float* __restrict__ w,
                                    int8_t* __restrict__ wq) {
  int idx = blockIdx.x * 256 + threadIdx.x;
  const int total = MTILES * KTILES * 32;
  if (idx >= total) return;
  int lane = idx & 31;
  int f    = idx >> 5;
  int mt   = f / KTILES;
  int kt   = f % KTILES;
  int m    = mt * 16 + (lane & 15);
  int k0   = kt * 64;
  int kbase = (lane < 16) ? 0 : 16;

  int frag[8];
  for (int v = 0; v < 8; ++v) {
    int koff = ((v >> 2) << 5) + ((v & 3) << 2) + kbase;
    uint32_t packed = 0;
    for (int d = 0; d < 4; ++d) {
      int c = k0 + koff + d;
      float wv = w[m * IN_CH + c];
      int8_t s = (int8_t)((wv > 0.0f) - (wv < 0.0f));
      packed |= ((uint32_t)(uint8_t)s) << (8 * d);
    }
    frag[v] = (int)packed;
  }
  v8i* dst = (v8i*)(wq + (size_t)idx * 32);
  v8i out;
  for (int v = 0; v < 8; ++v) out[v] = frag[v];
  *dst = out;
}

#define GT_STRIDE 400

__global__ void __launch_bounds__(256)
cyclefc1w1a_kernel(const float* __restrict__ x,
                   const int8_t* __restrict__ wq,
                   const float* __restrict__ bias,
                   float* __restrict__ out) {
  __shared__ __align__(16) int8_t gT[HW * GT_STRIDE];

  const int tid  = threadIdx.x;
  const int bi   = blockIdx.x;
  const int b    = bi >> 5;
  const int h    = bi & 31;

  const float* xrow = x + (((size_t)b * IN_CH) * HW + h) * HW;

  for (int j = tid; j < 96 * HW; j += 256) {
    int cg = j >> 5;
    int w  = j & 31;
    uint32_t packed = 0;
    for (int u = 0; u < 4; ++u) {
      int c   = cg * 4 + u;
      int off = (c + 3) % 7 - 3;
      int sc  = w + off;
      int8_t v = 0;
      if (sc >= 0 && sc < HW) {
        float xv = xrow[(size_t)c * (HW * HW) + sc];
        v = (int8_t)((xv > 0.0f) - (xv < 0.0f));
      }
      packed |= ((uint32_t)(uint8_t)v) << (8 * u);
    }
    *(uint32_t*)&gT[w * GT_STRIDE + cg * 4] = packed;
  }
  __syncthreads();

  const int wave = tid >> 5;
  const int lane = tid & 31;
  const int nlo  = lane & 15;

  for (int i = 0; i < 6; ++i) {
    int t  = wave * 6 + i;
    int mt = t >> 1;
    int nt = t & 1;
    int n  = nt * 16 + nlo;

    v8i acc = {};
    for (int kt = 0; kt < KTILES; ++kt) {
      const v8i* ap =
          (const v8i*)(wq + (((size_t)(mt * KTILES + kt) * 32) + lane) * 32);
      v8i A = *ap;

      const int8_t* bp =
          &gT[n * GT_STRIDE + kt * 64 + ((lane < 16) ? 0 : 16)];
      v4i b0 = *(const v4i*)bp;
      v4i b1 = *(const v4i*)(bp + 32);
      v8i B;
      B[0] = b0[0]; B[1] = b0[1]; B[2] = b0[2]; B[3] = b0[3];
      B[4] = b1[0]; B[5] = b1[1]; B[6] = b1[2]; B[7] = b1[3];

      acc = __builtin_amdgcn_wmma_i32_16x16x64_iu8(
           true, A,  true, B, acc,
           false,  false);
    }

    int mbase = mt * 16 + ((lane < 16) ? 0 : 8);
    float* orow = out + (((size_t)b * IN_CH + mbase) * HW + h) * HW + n;
    for (int r = 0; r < 8; ++r) {
      orow[(size_t)r * (HW * HW)] = (float)acc[r] + bias[mbase + r];
    }
  }
}

extern "C" void kernel_launch(void* const* d_in, const int* in_sizes, int n_in,
                              void* d_out, int out_size, void* d_ws, size_t ws_size,
                              hipStream_t stream) {
  const float* x      = (const float*)d_in[0];
  const float* weight = (const float*)d_in[1];
  const float* bias   = (const float*)d_in[2];
  float* out          = (float*)d_out;
  int8_t* wq          = (int8_t*)d_ws;

  quant_weight_kernel<<<(MTILES * KTILES * 32 + 255) / 256, 256, 0, stream>>>(
      weight, wq);

  cyclefc1w1a_kernel<<<NB * HW, 256, 0, stream>>>(x, wq, bias, out);
}
